// GraphGRU_12721693130980
// MI455X (gfx1250) — matrix operand order repaired and checked by reading, not compiled, not run on hardware
//
#include <hip/hip_runtime.h>

#define M_MSG 100000
#define NEI 8
#define HID 128
#define BM 16            // messages per block
#define ROWPAD 132       // 128 + 4 f32 pad (keeps 16B alignment, rotates banks)

typedef __bf16 bf16;
typedef bf16  v16bf __attribute__((ext_vector_type(16)));
typedef bf16  v8bf  __attribute__((ext_vector_type(8)));
typedef float v8f   __attribute__((ext_vector_type(8)));
typedef float v4f   __attribute__((ext_vector_type(4)));
typedef unsigned short u16;
typedef u16   v4u16 __attribute__((ext_vector_type(4)));

__device__ __forceinline__ bf16 f2bf(float f) {
    unsigned u = __builtin_bit_cast(unsigned, f);
    unsigned short s = (unsigned short)((u + 0x7FFFu + ((u >> 16) & 1u)) >> 16);
    return __builtin_bit_cast(bf16, s);
}

__device__ __forceinline__ v8f zerov8() {
    v8f z;
#pragma unroll
    for (int i = 0; i < 8; ++i) z[i] = 0.f;
    return z;
}

__device__ __forceinline__ v8f wmma_bf16(v16bf a, v16bf b, v8f c) {
    // D = A(16x32 bf16) * B(32x16 bf16) + C(16x16 f32)
    return __builtin_amdgcn_wmma_f32_16x16x32_bf16(false, a, false, b, (short)0, c,
                                                   false, false);
}

// A-fragment (16-bit A 16x32): lane half h picks contiguous K
// {k0+16h .. +15} (matches B). Source row is f32 in LDS.
__device__ __forceinline__ v16bf a_frag_f32(const float* rowp, int k0, int h) {
    const float* q0 = rowp + k0 + 16 * h;
    const float* q1 = q0 + 8;
    v4f x0 = *(const v4f*)(q0);
    v4f x1 = *(const v4f*)(q0 + 4);
    v4f y0 = *(const v4f*)(q1);
    v4f y1 = *(const v4f*)(q1 + 4);
    v16bf a;
#pragma unroll
    for (int i = 0; i < 4; ++i) {
        a[i]      = f2bf(x0[i]);
        a[4 + i]  = f2bf(x1[i]);
        a[8 + i]  = f2bf(y0[i]);
        a[12 + i] = f2bf(y1[i]);
    }
    return a;
}

// Same A-fragment but sourced from a pre-converted bf16 row in global memory.
__device__ __forceinline__ v16bf a_frag_bf16(const bf16* rowp, int k0, int h) {
    v8bf lo = *(const v8bf*)(rowp + k0 + 16 * h);
    v8bf hi = *(const v8bf*)(rowp + k0 + 16 * h + 8);
    v16bf a;
#pragma unroll
    for (int i = 0; i < 8; ++i) {
        a[i]     = lo[i];
        a[8 + i] = hi[i];
    }
    return a;
}

__device__ __forceinline__ float sigmoidf_(float x) {
    return 1.0f / (1.0f + __expf(-x));
}

// ---------------------------------------------------------------- cvt f32->bf16
__global__ __launch_bounds__(256) void cvt_f32_bf16(const float* __restrict__ src,
                                                    u16* __restrict__ dst, int n4) {
    int i = blockIdx.x * 256 + threadIdx.x;
    if (i >= n4) return;
    v4f x = *(const v4f*)(src + (size_t)i * 4);
    v4u16 y;
#pragma unroll
    for (int k = 0; k < 4; ++k) {
        unsigned u = __builtin_bit_cast(unsigned, x[k]);
        y[k] = (u16)((u + 0x7FFFu + ((u >> 16) & 1u)) >> 16);
    }
    *(v4u16*)(dst + (size_t)i * 4) = y;
}

// ---------------------------------------------------------------- r1 = lf @ W_r^T
__global__ __launch_bounds__(256) void r1_kernel(const bf16* __restrict__ lf_bf,
                                                 const bf16* __restrict__ Wr_bf,
                                                 float* __restrict__ r1g) {
    const int tid  = threadIdx.x;
    const int m0   = blockIdx.x * BM;
    const int lane = tid & 31;
    const int w    = tid >> 5;        // wave 0..7 -> col tile
    const int h    = (lane >> 4) & 1;
    const int ln   = lane & 15;

    const bf16* lfrow = lf_bf + (size_t)(m0 + ln) * HID;
    v8f c = zerov8();
#pragma unroll
    for (int ks = 0; ks < 4; ++ks) {
        v16bf a = a_frag_bf16(lfrow, ks * 32, h);
        // B[k][n] = W_r[n][k]; lane holds col n = 16w+ln, K = k0+16h .. +15
        v16bf b = *(const v16bf*)(Wr_bf + (size_t)(w * 16 + ln) * HID + ks * 32 + 16 * h);
        c = wmma_bf16(a, b, c);
    }
    const int col = w * 16 + ln;
#pragma unroll
    for (int v = 0; v < 8; ++v) {
        r1g[(size_t)(m0 + 8 * h + v) * HID + col] = c[v];
    }
}

// ---------------------------------------------------------------- one GRU depth step
__global__ __launch_bounds__(256, 2) void gru_step(
    const float* __restrict__ msg_in, float* __restrict__ msg_out,
    const bf16* __restrict__ lf_bf, const int* __restrict__ graph,
    const bf16* __restrict__ Wz_bf, const float* __restrict__ bz,
    const bf16* __restrict__ Ur_bf, const float* __restrict__ bur,
    const bf16* __restrict__ Wh_bf, const float* __restrict__ bh,
    const float* __restrict__ r1g)
{
    extern __shared__ char smem_raw[];
    float* s_msg = (float*)smem_raw;          // [128][ROWPAD] gathered neighbors (f32)
    float* s_r1  = s_msg + 128 * ROWPAD;      // [16][ROWPAD]
    float* s_sum = s_r1 + BM * ROWPAD;        // [16][ROWPAD] neighbor sums
    float* s_gat = s_sum + BM * ROWPAD;       // [16][ROWPAD] gated sums
    int*   s_idx = (int*)(s_gat + BM * ROWPAD); // [128]

    const int tid  = threadIdx.x;
    const int m0   = blockIdx.x * BM;
    const int lane = tid & 31;
    const int w    = tid >> 5;
    const int h    = (lane >> 4) & 1;
    const int ln   = lane & 15;

    // ---- Phase A1: neighbor indices + r1 tile into LDS
    if (tid < 128) s_idx[tid] = graph[m0 * NEI + tid];
    for (int i = tid; i < BM * 32; i += 256) {   // 16 rows x 32 float4
        int row = i >> 5, c4 = i & 31;
        v4f v = *(const v4f*)(r1g + (size_t)(m0 + row) * HID + c4 * 4);
        *(v4f*)(s_r1 + row * ROWPAD + c4 * 4) = v;
    }
    __syncthreads();

    // ---- Phase A2: gather msg_nei (128 rows x 128 f32) into LDS
    for (int i = tid; i < 128 * 32; i += 256) {
        int row = i >> 5, c4 = i & 31;
        int g = s_idx[row];
        v4f v = *(const v4f*)(msg_in + (size_t)g * HID + c4 * 4);
        *(v4f*)(s_msg + row * ROWPAD + c4 * 4) = v;
    }
    __syncthreads();

    // ---- Phase A3: sum over 8 neighbors
    for (int i = tid; i < BM * HID; i += 256) {
        int m = i >> 7, c = i & 127;
        float s = 0.f;
#pragma unroll
        for (int j = 0; j < NEI; ++j) s += s_msg[(m * NEI + j) * ROWPAD + c];
        s_sum[m * ROWPAD + c] = s;
    }
    __syncthreads();

    // ---- GEMM1: r2 = msg_nei @ U_r^T ; wave w owns row-tile w (rows 16w..16w+15)
    v16bf afr[4];
    {
        const float* rowp = s_msg + (w * 16 + ln) * ROWPAD;   // A row M = ln
#pragma unroll
        for (int ks = 0; ks < 4; ++ks) afr[ks] = a_frag_f32(rowp, ks * 32, h);
    }
    const int mloc = 2 * w + h;                         // message this lane reduces
    const float* s_msgrows = s_msg + (16 * w + 8 * h) * ROWPAD; // rows v=0..7
    const float  r1base = 0.f; (void)r1base;
#pragma unroll
    for (int ct = 0; ct < 8; ++ct) {
        const int col = ct * 16 + ln;
        v8f c = zerov8();
#pragma unroll
        for (int ks = 0; ks < 4; ++ks) {
            v16bf b = *(const v16bf*)(Ur_bf + (size_t)col * HID + ks * 32 + 16 * h);
            c = wmma_bf16(afr[ks], b, c);
        }
        // r = sigmoid(r1 + r2 + b_ur); gated sum over this message's 8 neighbors
        const float r1v = s_r1[mloc * ROWPAD + col];
        const float buv = bur[col];
        float acc = 0.f;
#pragma unroll
        for (int v = 0; v < 8; ++v) {
            float r = sigmoidf_(r1v + c[v] + buv);
            acc += r * s_msgrows[v * ROWPAD + col];
        }
        s_gat[mloc * ROWPAD + col] = acc;
    }
    __syncthreads();

    // ---- GEMM2 (z) and GEMM3 (h): wave w owns col-tile w, K = 256
    const bf16*  lfrow  = lf_bf + (size_t)(m0 + ln) * HID;   // A row M = ln
    const float* sumrow = s_sum + ln * ROWPAD;
    const float* gatrow = s_gat + ln * ROWPAD;

    v8f cz = zerov8();
#pragma unroll
    for (int ks = 0; ks < 8; ++ks) {
        v16bf a = (ks < 4) ? a_frag_bf16(lfrow, ks * 32, h)
                           : a_frag_f32(sumrow, ks * 32 - 128, h);
        v16bf b = *(const v16bf*)(Wz_bf + (size_t)(w * 16 + ln) * 256 + ks * 32 + 16 * h);
        cz = wmma_bf16(a, b, cz);
    }
    v8f ch = zerov8();
#pragma unroll
    for (int ks = 0; ks < 8; ++ks) {
        v16bf a = (ks < 4) ? a_frag_bf16(lfrow, ks * 32, h)
                           : a_frag_f32(gatrow, ks * 32 - 128, h);
        v16bf b = *(const v16bf*)(Wh_bf + (size_t)(w * 16 + ln) * 256 + ks * 32 + 16 * h);
        ch = wmma_bf16(a, b, ch);
    }

    // ---- Fused update: out = ((1-z)*sum_msg + z*tanh(h)) * mask
    const int col = w * 16 + ln;
    const float bzv = bz[col];
    const float bhv = bh[col];
#pragma unroll
    for (int v = 0; v < 8; ++v) {
        const int m = 8 * h + v;                  // C/D layout: row = 8h+v
        float z   = sigmoidf_(cz[v] + bzv);
        float pre = tanhf(ch[v] + bhv);
        float sm  = s_sum[m * ROWPAD + col];
        float out = (1.f - z) * sm + z * pre;
        const int grow = m0 + m;
        if (grow == 0) out = 0.f;                 // padding-row mask
        msg_out[(size_t)grow * HID + col] = out;
    }
}

// ----------------------------------------------------------------------------
extern "C" void kernel_launch(void* const* d_in, const int* in_sizes, int n_in,
                              void* d_out, int out_size, void* d_ws, size_t ws_size,
                              hipStream_t stream) {
    (void)in_sizes; (void)n_in; (void)out_size; (void)ws_size;

    const float* messages = (const float*)d_in[0];
    const int*   graph    = (const int*)d_in[2];
    const float* Wz       = (const float*)d_in[3];
    const float* bz       = (const float*)d_in[4];
    const float* Wr       = (const float*)d_in[5];
    const float* Ur       = (const float*)d_in[6];
    const float* bur      = (const float*)d_in[7];
    const float* Wh       = (const float*)d_in[8];
    const float* bh       = (const float*)d_in[9];
    const float* lf       = (const float*)d_in[1];
    float* out = (float*)d_out;

    char* p = (char*)d_ws;
    size_t off = 0;
    auto take = [&](size_t bytes) -> char* {
        char* r = p + off;
        off += (bytes + 255) & ~(size_t)255;
        return r;
    };
    float* msg_pong = (float*)take((size_t)M_MSG * HID * 4);
    float* r1g      = (float*)take((size_t)M_MSG * HID * 4);
    bf16*  lf_bf    = (bf16*)take((size_t)M_MSG * HID * 2);
    bf16*  Wz_bf    = (bf16*)take((size_t)HID * 256 * 2);
    bf16*  Wh_bf    = (bf16*)take((size_t)HID * 256 * 2);
    bf16*  Ur_bf    = (bf16*)take((size_t)HID * HID * 2);
    bf16*  Wr_bf    = (bf16*)take((size_t)HID * HID * 2);

    const size_t SMEM = (size_t)(128 * ROWPAD + 3 * BM * ROWPAD) * 4 + 128 * 4;
    hipFuncSetAttribute(reinterpret_cast<const void*>(gru_step),
                        hipFuncAttributeMaxDynamicSharedMemorySize, (int)SMEM);

    // One-time conversions (re-run every call for determinism; tiny cost)
    {
        int n4 = M_MSG * HID / 4;
        cvt_f32_bf16<<<(n4 + 255) / 256, 256, 0, stream>>>(lf, (u16*)lf_bf, n4);
        n4 = HID * 256 / 4;
        cvt_f32_bf16<<<(n4 + 255) / 256, 256, 0, stream>>>(Wz, (u16*)Wz_bf, n4);
        cvt_f32_bf16<<<(n4 + 255) / 256, 256, 0, stream>>>(Wh, (u16*)Wh_bf, n4);
        n4 = HID * HID / 4;
        cvt_f32_bf16<<<(n4 + 255) / 256, 256, 0, stream>>>(Ur, (u16*)Ur_bf, n4);
        cvt_f32_bf16<<<(n4 + 255) / 256, 256, 0, stream>>>(Wr, (u16*)Wr_bf, n4);
    }

    const int nblk = M_MSG / BM;   // 6250
    r1_kernel<<<nblk, 256, 0, stream>>>(lf_bf, Wr_bf, r1g);

    // DEPTH = 4, ping-pong: in -> ws -> out -> ws -> out
    gru_step<<<nblk, 256, SMEM, stream>>>(messages, msg_pong, lf_bf, graph,
                                          Wz_bf, bz, Ur_bf, bur, Wh_bf, bh, r1g);
    gru_step<<<nblk, 256, SMEM, stream>>>(msg_pong, out, lf_bf, graph,
                                          Wz_bf, bz, Ur_bf, bur, Wh_bf, bh, r1g);
    gru_step<<<nblk, 256, SMEM, stream>>>(out, msg_pong, lf_bf, graph,
                                          Wz_bf, bz, Ur_bf, bur, Wh_bf, bh, r1g);
    gru_step<<<nblk, 256, SMEM, stream>>>(msg_pong, out, lf_bf, graph,
                                          Wz_bf, bz, Ur_bf, bur, Wh_bf, bh, r1g);
}
